// Cate3Classifier_8065948582254
// MI455X (gfx1250) — matrix operand order repaired and checked by reading, compiled, not run on hardware
//
#include <hip/hip_runtime.h>

// ---------------------------------------------------------------------------
// Types
// ---------------------------------------------------------------------------
typedef __bf16 bf16_t;
typedef bf16_t v16bf __attribute__((ext_vector_type(16)));
typedef float  v8f   __attribute__((ext_vector_type(8)));
typedef unsigned short v8us  __attribute__((ext_vector_type(8)));
typedef unsigned short v16us __attribute__((ext_vector_type(16)));

#define B_ROWS 16384
#define D_IN   2048
#define H_D    512
#define C3     125
#define C3P    128
#define BN_EPS 1e-5f
#define MASKV  -100.0f
#define LDT    40   // LDS row stride in bf16 units (80B, 16B aligned)

// fp32 -> bf16 (round to nearest even), as raw ushort bits
__device__ __forceinline__ unsigned short f2bf(float f) {
    unsigned u = __builtin_bit_cast(unsigned, f);
    u += 0x7fffu + ((u >> 16) & 1u);
    return (unsigned short)(u >> 16);
}

// Build a v16bf WMMA fragment from two 16B LDS chunks
__device__ __forceinline__ v16bf ld_frag(const unsigned short* p0,
                                         const unsigned short* p1) {
    v8us a = *(const v8us*)p0;
    v8us b = *(const v8us*)p1;
    v16us c = __builtin_shufflevector(a, b, 0,1,2,3,4,5,6,7,8,9,10,11,12,13,14,15);
    return __builtin_bit_cast(v16bf, c);
}

__device__ __forceinline__ v8f wmma_bf16(v16bf a, v16bf b, v8f c) {
#if defined(__gfx1250__)
    return __builtin_amdgcn_wmma_f32_16x16x32_bf16(false, a, false, b,
                                                   (short)0, c, false, false);
#else
    return c;
#endif
}

// Async DMA: 16 bytes global -> LDS, tracked by ASYNCcnt (CDNA5 path)
__device__ __forceinline__ void async_copy_b128(unsigned lds_off, const void* gptr) {
#if defined(__gfx1250__)
    asm volatile("global_load_async_to_lds_b128 %0, %1, off"
                 :: "v"(lds_off),
                    "v"((unsigned long long)(uintptr_t)gptr)
                 : "memory");
#else
    (void)lds_off; (void)gptr;
#endif
}

__device__ __forceinline__ void wait_async() {
#if defined(__gfx1250__)
    asm volatile("s_wait_asynccnt 0x0" ::: "memory");
#endif
}

// ---------------------------------------------------------------------------
// Kernel: zero BN accumulators (1024 floats: sum[512] ++ sumsq[512])
// ---------------------------------------------------------------------------
__global__ void zero_stats_k(float* p) {
    int i = blockIdx.x * blockDim.x + threadIdx.x;
    if (i < 1024) p[i] = 0.0f;
}

// ---------------------------------------------------------------------------
// Kernel: convert + transpose weights to bf16
//   WfcT[n*2048 + k]  = bf16(W_fc[k*512 + n])              (512 x 2048)
//   WclfT[n*512 + k]  = n<125 ? bf16(W_clf[k*125+n]) : 0   (128 x 512)
// ---------------------------------------------------------------------------
__global__ void cvt_weights_k(const float* __restrict__ Wfc,
                              const float* __restrict__ Wclf,
                              unsigned short* __restrict__ WfcT,
                              unsigned short* __restrict__ WclfT) {
    int idx = blockIdx.x * blockDim.x + threadIdx.x;
    int stride = gridDim.x * blockDim.x;
    for (int i = idx; i < D_IN * H_D; i += stride) {
        int k = i / H_D, n = i % H_D;
        WfcT[n * D_IN + k] = f2bf(Wfc[i]);
    }
    for (int i = idx; i < C3P * H_D; i += stride) {
        int n = i / H_D, k = i % H_D;
        WclfT[i] = (n < C3) ? f2bf(Wclf[k * C3 + n]) : (unsigned short)0;
    }
}

// ---------------------------------------------------------------------------
// Kernel: GEMM1 (h = X @ W_fc + b_fc) + BN partial stats
//   grid = (4 N-blocks, 128 M-blocks), block = 256 (8 waves, 4M x 2N)
//   block tile 128x128, wave tile 32x64 (2x4 WMMA frags), K step 32
//   double-buffered LDS; B tile staged via async global->LDS DMA
// ---------------------------------------------------------------------------
__global__ void __launch_bounds__(256)
gemm1_bn_k(const float* __restrict__ X,
           const unsigned short* __restrict__ WfcT,
           const float* __restrict__ bfc,
           float* __restrict__ h,
           float* __restrict__ gsum,
           float* __restrict__ gsq) {
    __shared__ unsigned short Xs[2][128 * LDT];
    __shared__ unsigned short Ws[2][128 * LDT];
    __shared__ float lsum[128];
    __shared__ float lsq[128];

    const int t    = threadIdx.x;
    const int lane = t & 31;
    const int wave = t >> 5;
    const int wm   = wave >> 1;   // 0..3
    const int wn   = wave & 1;    // 0..1
    const int l15  = lane & 15;
    const int hi   = lane >> 4;   // 0/1
    const int rowG0 = blockIdx.y * 128;
    const int colG0 = blockIdx.x * 128;

    if (t < 128) { lsum[t] = 0.0f; lsq[t] = 0.0f; }

    v8f acc[2][4] = {};

    // staging assignments
    const int xr = t >> 1;            // 0..127
    const int xk = (t & 1) * 16;      // 0/16
    const float* xsrc = X + (size_t)(rowG0 + xr) * D_IN + xk;
    const int wcol = t >> 2;          // 0..63 (+64 for second chunk)
    const int wseg = (t & 3) * 8;     // 0/8/16/24 (bf16 units)

    auto fill = [&](int buf, int k0) {
        // A tile: fp32 -> bf16 conversion, 16 values per thread
        const float4* f4 = (const float4*)(xsrc + k0);
        float4 a0 = f4[0], a1 = f4[1], a2 = f4[2], a3 = f4[3];
        v8us p0, p1;
        p0[0]=f2bf(a0.x); p0[1]=f2bf(a0.y); p0[2]=f2bf(a0.z); p0[3]=f2bf(a0.w);
        p0[4]=f2bf(a1.x); p0[5]=f2bf(a1.y); p0[6]=f2bf(a1.z); p0[7]=f2bf(a1.w);
        p1[0]=f2bf(a2.x); p1[1]=f2bf(a2.y); p1[2]=f2bf(a2.z); p1[3]=f2bf(a2.w);
        p1[4]=f2bf(a3.x); p1[5]=f2bf(a3.y); p1[6]=f2bf(a3.z); p1[7]=f2bf(a3.w);
        unsigned short* xdst = &Xs[buf][xr * LDT + xk];
        *(v8us*)(xdst)     = p0;
        *(v8us*)(xdst + 8) = p1;
        // B tile: 2x 16B async DMA per thread (128 cols x 32 bf16)
        async_copy_b128((unsigned)(uintptr_t)&Ws[buf][wcol * LDT + wseg],
                        WfcT + (size_t)(colG0 + wcol) * D_IN + k0 + wseg);
        async_copy_b128((unsigned)(uintptr_t)&Ws[buf][(wcol + 64) * LDT + wseg],
                        WfcT + (size_t)(colG0 + wcol + 64) * D_IN + k0 + wseg);
    };

    fill(0, 0);
    const int NK = D_IN / 32;   // 64
    for (int kt = 0; kt < NK; ++kt) {
        wait_async();
        __syncthreads();
        const int cur = kt & 1;
        if (kt + 1 < NK) fill(cur ^ 1, (kt + 1) * 32);

        v16bf aF[2], bF[4];
#pragma unroll
        for (int mi = 0; mi < 2; ++mi) {
            const unsigned short* pa = &Xs[cur][(wm * 32 + mi * 16 + l15) * LDT + hi * 8];
            aF[mi] = ld_frag(pa, pa + 16);
        }
#pragma unroll
        for (int ni = 0; ni < 4; ++ni) {
            const unsigned short* pb = &Ws[cur][(wn * 64 + ni * 16 + l15) * LDT + hi * 8];
            bF[ni] = ld_frag(pb, pb + 16);
        }
#pragma unroll
        for (int mi = 0; mi < 2; ++mi)
#pragma unroll
            for (int ni = 0; ni < 4; ++ni)
                acc[mi][ni] = wmma_bf16(aF[mi], bF[ni], acc[mi][ni]);
    }

    // epilogue: h = acc + bias; BN partial sums via LDS atomics
#pragma unroll
    for (int ni = 0; ni < 4; ++ni) {
        const int colL = wn * 64 + ni * 16 + l15;
        const int colG = colG0 + colL;
        const float bias = bfc[colG];
        float s = 0.0f, q = 0.0f;
#pragma unroll
        for (int mi = 0; mi < 2; ++mi) {
#pragma unroll
            for (int i = 0; i < 8; ++i) {
                float v = acc[mi][ni][i] + bias;
                int rowG = rowG0 + wm * 32 + mi * 16 + hi * 8 + i;
                h[(size_t)rowG * H_D + colG] = v;
                s += v;
                q += v * v;
            }
        }
        atomicAdd(&lsum[colL], s);
        atomicAdd(&lsq[colL], q);
    }
    __syncthreads();
    if (t < 128) {
        atomicAdd(&gsum[colG0 + t], lsum[t]);
        atomicAdd(&gsq[colG0 + t], lsq[t]);
    }
}

// ---------------------------------------------------------------------------
// Kernel: finalize BN -> scale/shift
// ---------------------------------------------------------------------------
__global__ void finalize_bn_k(const float* __restrict__ gsum,
                              const float* __restrict__ gsq,
                              const float* __restrict__ gamma,
                              const float* __restrict__ beta,
                              float* __restrict__ scale,
                              float* __restrict__ shiftv) {
    int t = blockIdx.x * blockDim.x + threadIdx.x;
    if (t < H_D) {
        const float invB = 1.0f / (float)B_ROWS;
        float mu  = gsum[t] * invB;
        float var = gsq[t] * invB - mu * mu;
        float inv = rsqrtf(var + BN_EPS);
        float sc  = gamma[t] * inv;
        scale[t]  = sc;
        shiftv[t] = beta[t] - mu * sc;
    }
}

// ---------------------------------------------------------------------------
// Kernel: GEMM2 (out = relu(bn(h)) @ W_clf + b_clf) + ragged mask
//   grid = 128 M-blocks, block = 256 (8 waves, 4M x 2N)
//   block tile 128x128 (N=125 padded to 128), wave tile 32x64, K step 32
//   double-buffered LDS; B tile staged via async global->LDS DMA
// ---------------------------------------------------------------------------
__global__ void __launch_bounds__(256)
gemm2_mask_k(const float* __restrict__ hbuf,
             const unsigned short* __restrict__ WclfT,
             const float* __restrict__ scale,
             const float* __restrict__ shiftv,
             const float* __restrict__ bclf,
             const int* __restrict__ cate2,
             const unsigned int* __restrict__ mask2w,
             float* __restrict__ out) {
    __shared__ unsigned short Hs[2][128 * LDT];
    __shared__ unsigned short Ws[2][128 * LDT];
    __shared__ float sc[H_D];
    __shared__ float sh[H_D];
    __shared__ float bc[C3P];
    __shared__ unsigned int mwords[2000];   // 64*125 = 8000 bytes
    __shared__ int cat[128];

    const int t    = threadIdx.x;
    const int lane = t & 31;
    const int wave = t >> 5;
    const int wm   = wave >> 1;   // 0..3
    const int wn   = wave & 1;    // 0..1
    const int l15  = lane & 15;
    const int hi   = lane >> 4;
    const int rowG0 = blockIdx.x * 128;

    // prologue staging (sc/sh consumed by the A-tile fill -> barrier below)
    sc[t] = scale[t];         sc[t + 256] = scale[t + 256];
    sh[t] = shiftv[t];        sh[t + 256] = shiftv[t + 256];
    if (t < 128) cat[t] = cate2[rowG0 + t];
    if (t < C3)  bc[t]  = bclf[t];
    for (int i = t; i < 2000; i += 256) mwords[i] = mask2w[i];
    __syncthreads();

    v8f acc[2][4] = {};

    const int hr = t >> 1;            // 0..127
    const int hk = (t & 1) * 16;      // 0/16
    const float* hsrc = hbuf + (size_t)(rowG0 + hr) * H_D + hk;
    const int wcol = t >> 2;          // 0..63 (+64 for second chunk)
    const int wseg = (t & 3) * 8;     // 0/8/16/24

    auto fill = [&](int buf, int k0) {
        // A tile: bn + relu + bf16 on the fly (16 values per thread)
        const float4* f4 = (const float4*)(hsrc + k0);
        float vals[16];
        *(float4*)&vals[0]  = f4[0];
        *(float4*)&vals[4]  = f4[1];
        *(float4*)&vals[8]  = f4[2];
        *(float4*)&vals[12] = f4[3];
        v8us p0, p1;
#pragma unroll
        for (int j = 0; j < 16; ++j) {
            int k = k0 + hk + j;
            float x = fmaxf(vals[j] * sc[k] + sh[k], 0.0f);
            if (j < 8) p0[j] = f2bf(x); else p1[j - 8] = f2bf(x);
        }
        unsigned short* hdst = &Hs[buf][hr * LDT + hk];
        *(v8us*)(hdst)     = p0;
        *(v8us*)(hdst + 8) = p1;
        // B tile: 2x 16B async DMA per thread
        async_copy_b128((unsigned)(uintptr_t)&Ws[buf][wcol * LDT + wseg],
                        WclfT + (size_t)wcol * H_D + k0 + wseg);
        async_copy_b128((unsigned)(uintptr_t)&Ws[buf][(wcol + 64) * LDT + wseg],
                        WclfT + (size_t)(wcol + 64) * H_D + k0 + wseg);
    };

    fill(0, 0);
    const int NK = H_D / 32;   // 16
    for (int kt = 0; kt < NK; ++kt) {
        wait_async();
        __syncthreads();
        const int cur = kt & 1;
        if (kt + 1 < NK) fill(cur ^ 1, (kt + 1) * 32);

        v16bf aF[2], bF[4];
#pragma unroll
        for (int mi = 0; mi < 2; ++mi) {
            const unsigned short* pa = &Hs[cur][(wm * 32 + mi * 16 + l15) * LDT + hi * 8];
            aF[mi] = ld_frag(pa, pa + 16);
        }
#pragma unroll
        for (int ni = 0; ni < 4; ++ni) {
            const unsigned short* pb = &Ws[cur][(wn * 64 + ni * 16 + l15) * LDT + hi * 8];
            bF[ni] = ld_frag(pb, pb + 16);
        }
#pragma unroll
        for (int mi = 0; mi < 2; ++mi)
#pragma unroll
            for (int ni = 0; ni < 4; ++ni)
                acc[mi][ni] = wmma_bf16(aF[mi], bF[ni], acc[mi][ni]);
    }

    // epilogue: bias + ragged mask + store (cols >= 125 dropped)
    const unsigned char* mbytes = (const unsigned char*)mwords;
#pragma unroll
    for (int ni = 0; ni < 4; ++ni) {
        const int col = wn * 64 + ni * 16 + l15;
        if (col < C3) {
            const float bias = bc[col];
#pragma unroll
            for (int mi = 0; mi < 2; ++mi) {
#pragma unroll
                for (int i = 0; i < 8; ++i) {
                    const int rowL = wm * 32 + mi * 16 + hi * 8 + i;
                    const int rowG = rowG0 + rowL;
                    float v = acc[mi][ni][i] + bias;
                    unsigned char mb = mbytes[cat[rowL] * C3 + col];
                    out[(size_t)rowG * C3 + col] = mb ? MASKV : v;
                }
            }
        }
    }
}

// ---------------------------------------------------------------------------
// Host launcher
// ---------------------------------------------------------------------------
extern "C" void kernel_launch(void* const* d_in, const int* in_sizes, int n_in,
                              void* d_out, int out_size, void* d_ws, size_t ws_size,
                              hipStream_t stream) {
    (void)in_sizes; (void)n_in; (void)out_size; (void)ws_size;

    const float*        X     = (const float*)d_in[0];
    const int*          cate2 = (const int*)d_in[1];
    const unsigned int* mask2 = (const unsigned int*)d_in[2]; // bool[64][125] as bytes
    const float*        Wfc   = (const float*)d_in[3];
    const float*        bfc   = (const float*)d_in[4];
    const float*        gamma = (const float*)d_in[5];
    const float*        beta  = (const float*)d_in[6];
    const float*        Wclf  = (const float*)d_in[7];
    const float*        bclf  = (const float*)d_in[8];
    float*              out   = (float*)d_out;

    char* ws = (char*)d_ws;
    // workspace layout (bytes)
    float*          h     = (float*)(ws);                               // 16384*512*4 = 33554432
    unsigned short* WfcT  = (unsigned short*)(ws + 33554432);           // 512*2048*2  =  2097152
    unsigned short* WclfT = (unsigned short*)(ws + 35651584);           // 128*512*2   =   131072
    float*          gsum  = (float*)(ws + 35782656);                    // 512 floats
    float*          gsq   = gsum + 512;
    float*          scale = gsum + 1024;
    float*          shiftv= gsum + 1536;

    zero_stats_k<<<4, 256, 0, stream>>>(gsum);
    cvt_weights_k<<<512, 256, 0, stream>>>(Wfc, Wclf, WfcT, WclfT);

    dim3 grid1(H_D / 128, B_ROWS / 128);   // (4, 128)
    gemm1_bn_k<<<grid1, 256, 0, stream>>>(X, WfcT, bfc, h, gsum, gsq);

    finalize_bn_k<<<2, 256, 0, stream>>>(gsum, gsq, gamma, beta, scale, shiftv);

    gemm2_mask_k<<<B_ROWS / 128, 256, 0, stream>>>(h, WclfT, scale, shiftv,
                                                   bclf, cate2, mask2, out);
}
